// UnidirectionalAttention_23854248362844
// MI455X (gfx1250) — matrix operand order repaired and checked by reading, not compiled, not run on hardware
//
#include <hip/hip_runtime.h>
#include <hip/hip_bf16.h>
#include <stdint.h>

// ---------------- problem constants (from reference) ----------------
#define SEQ     2048
#define HID     1024
#define NHEADS  16
#define HD      64          // head dim
#define BATCH   4
#define MTOT    (BATCH*SEQ) // 8192 tokens
#define NEG_BIG -10000.0f

typedef _Float16 v16h __attribute__((ext_vector_type(16)));
typedef _Float16 v8h  __attribute__((ext_vector_type(8)));
typedef _Float16 v4h  __attribute__((ext_vector_type(4)));
typedef float    v8f  __attribute__((ext_vector_type(8)));
typedef float    v4f  __attribute__((ext_vector_type(4)));
typedef unsigned int u32x4 __attribute__((ext_vector_type(4)));
typedef int          i32x4 __attribute__((ext_vector_type(4)));
typedef int          i32x8 __attribute__((ext_vector_type(8)));

#if defined(__has_builtin)
#if __has_builtin(__builtin_amdgcn_tensor_load_to_lds) && \
    __has_builtin(__builtin_amdgcn_s_wait_tensorcnt)
#define HAVE_TDM 1
#endif
#endif
#ifndef HAVE_TDM
#define HAVE_TDM 0
#endif

// ---------------- WMMA helpers (gfx1250, wave32) ----------------
__device__ __forceinline__ v8f wmma_f16(v16h a, v16h b, v8f c) {
    // D = A(16x32 f16) * B(32x16 f16) + C(16x16 f32)
    return __builtin_amdgcn_wmma_f32_16x16x32_f16(
        false, a, false, b, (short)0, c, false, false);
}

// A fragment (16x32, M x K): lane = m + 16*kh; halves[0..7]  = A[m][8*kh .. +7]
//                                              halves[8..15] = A[m][16+8*kh .. +7]
__device__ __forceinline__ v16h load_a_frag(const _Float16* tile, int ld, int lane) {
    const int m  = lane & 15;
    const int kh = lane >> 4;
    const _Float16* r = tile + m * ld;
    v8h lo = *(const v8h*)(r + 8 * kh);
    v8h hi = *(const v8h*)(r + 16 + 8 * kh);
    v16h a;
#pragma unroll
    for (int i = 0; i < 8; ++i) { a[i] = lo[i]; a[i + 8] = hi[i]; }
    return a;
}

// B fragment (32x16, K x N) read from a K-transposed tile stored [n][k]:
// lane = n + 16*kh; halves[0..7] = Bt[n][8*kh .. +7], halves[8..15] = Bt[n][16+8*kh .. +7]
__device__ __forceinline__ v16h load_b_frag(const _Float16* tileT, int ld, int lane) {
    const int n  = lane & 15;
    const int kh = lane >> 4;
    const _Float16* r = tileT + n * ld + 8 * kh;
    v8h lo = *(const v8h*)(r);
    v8h hi = *(const v8h*)(r + 16);
    v16h b;
#pragma unroll
    for (int i = 0; i < 8; ++i) { b[i] = lo[i]; b[i + 8] = hi[i]; }
    return b;
}

// ---------------- TDM: async 2-D f16 tile load, global -> LDS ----------------
// D# per CDNA5 ISA. data_size=4B units. LDS padding is
// produced by the TDM itself (pad after every row) so fragment reads avoid
// bank conflicts without a separate repack.
#if HAVE_TDM
__device__ __forceinline__ void tdm_load_tile_f16(
        uint32_t lds_addr, const void* gptr,
        uint32_t tile_w_dw, uint32_t tile_h,
        uint32_t tensor_w_dw, uint32_t tensor_h, uint32_t stride_dw,
        uint32_t pad_interval, uint32_t pad_amount) {
    uint64_t ga = (uint64_t)(uintptr_t)gptr;
    u32x4 g0;
    g0[0] = 1u;                                   // count=1, user mode, load
    g0[1] = lds_addr;                             // LDS byte address
    g0[2] = (uint32_t)ga;                         // global addr [95:64]
    g0[3] = (uint32_t)((ga >> 32) & 0x1FFFFFFu)   // global addr [120:96]
          | 0x80000000u;                          // type=2 ("image") [127:126]
    i32x8 g1;
    g1[0] = (int)((2u << 16)                      // data_size = 4 bytes
          | (pad_amount ? (1u << 20) : 0u)        // pad_enable
          | (pad_interval << 22)                  // pad every 2^(pi+1) DWORDs
          | (pad_amount << 25));                  // pad (pa+1) DWORDs
    g1[1] = (int)((tensor_w_dw & 0xFFFFu) << 16);           // tensor_dim0 lo16
    g1[2] = (int)((tensor_w_dw >> 16) | ((tensor_h & 0xFFFFu) << 16));
    g1[3] = (int)((tensor_h >> 16) | (tile_w_dw << 16));    // tile_dim0
    g1[4] = (int)(tile_h & 0xFFFFu);                        // tile_dim1 (tile_dim2=0)
    g1[5] = (int)stride_dw;                                 // tensor_dim0_stride lo32
    g1[6] = 0;                                              // stride hi16, dim1_stride lo16
    g1[7] = 0;
    i32x4 z4 = {0, 0, 0, 0};
#if __clang_major__ >= 23
    i32x8 z8 = {0, 0, 0, 0, 0, 0, 0, 0};
    __builtin_amdgcn_tensor_load_to_lds(g0, g1, z4, z4, z8, 0);
#else
    __builtin_amdgcn_tensor_load_to_lds(g0, g1, z4, z4, 0);
#endif
}
#endif

// ---------------- kernel 0: transpose + f32->f16 convert (weights) ----------------
__global__ __launch_bounds__(256)
void k_transpose_cvt(const float* __restrict__ w, _Float16* __restrict__ wt,
                     int K, int N) {
    size_t idx = (size_t)blockIdx.x * 256 + threadIdx.x;
    size_t total = (size_t)K * N;
    if (idx >= total) return;
    int k = (int)(idx / N);
    int n = (int)(idx % N);
    wt[(size_t)n * K + k] = (_Float16)w[idx];
}

// ---------------- GEMM kernels ----------------
// 128x128 output tile / 256 threads (8 waves, each a 32x64 sub-tile),
// K-tile = 64 (2 WMMA K-steps, 16 WMMAs per wave per barrier round).
// B tile (f16 [n][k]) is DMA'd by the Tensor Data Mover, double buffered.
#define LDA 72            // 64 + 8 pad (halves); TDM pad: 32dw interval, 4dw amount
#define KT  64
#define NKT (HID / KT)    // 16 rounds

// kernel 1: C[8192x3072] = x(f32) @ qkv_wT(f16)^T + bias -> scatter Q/K/V f16
__global__ __launch_bounds__(256)
void k_qkv_gemm(const float* __restrict__ x, const _Float16* __restrict__ wT,
                const float* __restrict__ bias,
                _Float16* __restrict__ qh, _Float16* __restrict__ kkh,
                _Float16* __restrict__ vh) {
    __shared__ __align__(16) _Float16 As[128 * LDA];
    __shared__ __align__(16) _Float16 Bs[2][128 * LDA];

    const int tid  = threadIdx.x;
    const int lane = tid & 31;
    const int wv   = tid >> 5;
    const int n0 = blockIdx.x * 128;
    const int m0 = blockIdx.y * 128;
    const int wm = (wv >> 1) * 32;
    const int wn = (wv & 1) * 64;

    v8f acc[2][4];
#pragma unroll
    for (int i = 0; i < 2; ++i)
#pragma unroll
        for (int j = 0; j < 4; ++j)
#pragma unroll
            for (int e = 0; e < 8; ++e) acc[i][j][e] = 0.0f;

#if HAVE_TDM
    if (wv == 0)
        tdm_load_tile_f16((uint32_t)(uintptr_t)&Bs[0][0],
                          wT + (size_t)n0 * HID, /*tile_w_dw=*/KT / 2, /*tile_h=*/128,
                          /*tensor_w_dw=*/HID / 2, /*tensor_h=*/3 * HID,
                          /*stride_dw=*/HID / 2, /*pad_interval=*/4, /*pad_amount=*/3);
#endif

    for (int kt = 0; kt < NKT; ++kt) {
        const int k0 = kt * KT;
        __syncthreads();
        // A tile: 128x64 f32 -> f16 into LDS (8 chunks of 4 floats / thread)
#pragma unroll
        for (int i = 0; i < 8; ++i) {
            int c = tid + 256 * i;
            int r = c >> 4, col = (c & 15) * 4;
            v4f d = *(const v4f*)(x + (size_t)(m0 + r) * HID + k0 + col);
            v4h h4;
#pragma unroll
            for (int e = 0; e < 4; ++e) h4[e] = (_Float16)d[e];
            *(v4h*)(&As[r * LDA + col]) = h4;
        }
#if HAVE_TDM
        if (wv == 0) {
            if (kt + 1 < NKT) {
                tdm_load_tile_f16((uint32_t)(uintptr_t)&Bs[(kt + 1) & 1][0],
                                  wT + (size_t)n0 * HID + (k0 + KT),
                                  KT / 2, 128, HID / 2, 3 * HID, HID / 2, 4, 3);
                __builtin_amdgcn_s_wait_tensorcnt(1);
            } else {
                __builtin_amdgcn_s_wait_tensorcnt(0);
            }
        }
#else
        // fallback: manual B tile load (4 chunks of 8 halves / thread)
#pragma unroll
        for (int i = 0; i < 4; ++i) {
            int c = tid + 256 * i;
            int r = c >> 3, col = (c & 7) * 8;
            *(v8h*)(&Bs[kt & 1][r * LDA + col]) =
                *(const v8h*)(wT + (size_t)(n0 + r) * HID + k0 + col);
        }
#endif
        __syncthreads();

        const _Float16* Bcur = &Bs[kt & 1][0];
#pragma unroll
        for (int kw = 0; kw < 2; ++kw) {
            v16h af0 = load_a_frag(&As[wm * LDA + kw * 32], LDA, lane);
            v16h af1 = load_a_frag(&As[(wm + 16) * LDA + kw * 32], LDA, lane);
#pragma unroll
            for (int nj = 0; nj < 4; ++nj) {
                v16h bf = load_b_frag(&Bcur[(wn + nj * 16) * LDA + kw * 32], LDA, lane);
                acc[0][nj] = wmma_f16(af0, bf, acc[0][nj]);
                acc[1][nj] = wmma_f16(af1, bf, acc[1][nj]);
            }
        }
    }

    // epilogue: + bias, scatter into Q/K/V  [b][h][s][l] f16
    const int ln = lane & 15, lh = lane >> 4;
#pragma unroll
    for (int mi = 0; mi < 2; ++mi) {
#pragma unroll
        for (int nj = 0; nj < 4; ++nj) {
            int n = n0 + wn + nj * 16 + ln;
            float bv = bias[n];
            int which = n >> 10, f = n & 1023, h = f >> 6, l = f & 63;
            _Float16* dst = (which == 0) ? qh : (which == 1) ? kkh : vh;
#pragma unroll
            for (int j = 0; j < 8; ++j) {
                int m = m0 + wm + mi * 16 + j + 8 * lh;
                int b = m >> 11, s = m & 2047;
                dst[(((size_t)(b * NHEADS + h)) * SEQ + s) * HD + l] =
                    (_Float16)(acc[mi][nj][j] + bv);
            }
        }
    }
}

// kernel 3: out[8192x1024] = ctx(f16) @ out_wT(f16)^T + out_b (fp32 store)
__global__ __launch_bounds__(256)
void k_out_gemm(const _Float16* __restrict__ a, const _Float16* __restrict__ wT,
                const float* __restrict__ bias, float* __restrict__ out) {
    __shared__ __align__(16) _Float16 As[128 * LDA];
    __shared__ __align__(16) _Float16 Bs[2][128 * LDA];

    const int tid  = threadIdx.x;
    const int lane = tid & 31;
    const int wv   = tid >> 5;
    const int n0 = blockIdx.x * 128;
    const int m0 = blockIdx.y * 128;
    const int wm = (wv >> 1) * 32;
    const int wn = (wv & 1) * 64;

    v8f acc[2][4];
#pragma unroll
    for (int i = 0; i < 2; ++i)
#pragma unroll
        for (int j = 0; j < 4; ++j)
#pragma unroll
            for (int e = 0; e < 8; ++e) acc[i][j][e] = 0.0f;

#if HAVE_TDM
    if (wv == 0)
        tdm_load_tile_f16((uint32_t)(uintptr_t)&Bs[0][0],
                          wT + (size_t)n0 * HID, KT / 2, 128,
                          HID / 2, HID, HID / 2, 4, 3);
#endif

    for (int kt = 0; kt < NKT; ++kt) {
        const int k0 = kt * KT;
        __syncthreads();
        // A tile: 128x64 f16 (4 chunks of 8 halves / thread)
#pragma unroll
        for (int i = 0; i < 4; ++i) {
            int c = tid + 256 * i;
            int r = c >> 3, col = (c & 7) * 8;
            *(v8h*)(&As[r * LDA + col]) =
                *(const v8h*)(a + (size_t)(m0 + r) * HID + k0 + col);
        }
#if HAVE_TDM
        if (wv == 0) {
            if (kt + 1 < NKT) {
                tdm_load_tile_f16((uint32_t)(uintptr_t)&Bs[(kt + 1) & 1][0],
                                  wT + (size_t)n0 * HID + (k0 + KT),
                                  KT / 2, 128, HID / 2, HID, HID / 2, 4, 3);
                __builtin_amdgcn_s_wait_tensorcnt(1);
            } else {
                __builtin_amdgcn_s_wait_tensorcnt(0);
            }
        }
#else
#pragma unroll
        for (int i = 0; i < 4; ++i) {
            int c = tid + 256 * i;
            int r = c >> 3, col = (c & 7) * 8;
            *(v8h*)(&Bs[kt & 1][r * LDA + col]) =
                *(const v8h*)(wT + (size_t)(n0 + r) * HID + k0 + col);
        }
#endif
        __syncthreads();

        const _Float16* Bcur = &Bs[kt & 1][0];
#pragma unroll
        for (int kw = 0; kw < 2; ++kw) {
            v16h af0 = load_a_frag(&As[wm * LDA + kw * 32], LDA, lane);
            v16h af1 = load_a_frag(&As[(wm + 16) * LDA + kw * 32], LDA, lane);
#pragma unroll
            for (int nj = 0; nj < 4; ++nj) {
                v16h bf = load_b_frag(&Bcur[(wn + nj * 16) * LDA + kw * 32], LDA, lane);
                acc[0][nj] = wmma_f16(af0, bf, acc[0][nj]);
                acc[1][nj] = wmma_f16(af1, bf, acc[1][nj]);
            }
        }
    }

    const int ln = lane & 15, lh = lane >> 4;
#pragma unroll
    for (int mi = 0; mi < 2; ++mi) {
#pragma unroll
        for (int nj = 0; nj < 4; ++nj) {
            int n = n0 + wn + nj * 16 + ln;
            float bv = bias[n];
#pragma unroll
            for (int j = 0; j < 8; ++j) {
                int m = m0 + wm + mi * 16 + j + 8 * lh;
                out[(size_t)m * HID + n] = acc[mi][nj][j] + bv;
            }
        }
    }
}

// ---------------- kernel 2: flash attention (causal) ----------------
// grid: x = S/128 query blocks, y = B*NHEADS. 8 waves, wave w owns 16 query
// rows; 64-wide key blocks, online softmax, all matmuls via WMMA.
#define LQ 72   // 64 + 8 pad
__global__ __launch_bounds__(256)
void k_attn(const _Float16* __restrict__ qh, const _Float16* __restrict__ kkh,
            const _Float16* __restrict__ vh, _Float16* __restrict__ ctx) {
    __shared__ __align__(16) _Float16 Qs[128 * LQ];
    __shared__ __align__(16) _Float16 Ks[64 * LQ];
    __shared__ __align__(16) _Float16 Vt[64 * LQ];     // V transposed: [l][key]
    __shared__ __align__(16) _Float16 Ps[8 * 16 * LQ]; // per-wave P staging

    const int tid  = threadIdx.x;
    const int lane = tid & 31;
    const int wv   = tid >> 5;
    const int ln = lane & 15, lh = lane >> 4;
    const int qb = blockIdx.x;
    const int bz = blockIdx.y;                  // b*16 + h
    const size_t headBase = (size_t)bz * SEQ * HD;
    _Float16* Pw = &Ps[wv * 16 * LQ];

#pragma unroll
    for (int i = 0; i < 4; ++i) {
        int c = tid + 256 * i;
        int r = c >> 3, col = (c & 7) * 8;
        *(v8h*)(&Qs[r * LQ + col]) =
            *(const v8h*)(qh + headBase + (size_t)(qb * 128 + r) * HD + col);
    }

    float rM[8], rL[8];
    v8f o[4];
#pragma unroll
    for (int j = 0; j < 8; ++j) { rM[j] = -1e30f; rL[j] = 0.0f; }
#pragma unroll
    for (int t = 0; t < 4; ++t)
#pragma unroll
        for (int e = 0; e < 8; ++e) o[t][e] = 0.0f;

    const int qbase = qb * 128 + wv * 16;
    const int kbMax = 2 * qb + 1;

    for (int kb = 0; kb <= kbMax; ++kb) {
        __syncthreads();
#pragma unroll
        for (int i = 0; i < 2; ++i) {
            int c = tid + 256 * i;
            int r = c >> 3, col = (c & 7) * 8;
            *(v8h*)(&Ks[r * LQ + col]) =
                *(const v8h*)(kkh + headBase + (size_t)(kb * 64 + r) * HD + col);
        }
#pragma unroll
        for (int i = 0; i < 2; ++i) {
            int c = tid + 256 * i;
            int r = c >> 3, col = (c & 7) * 8;
            v8h d = *(const v8h*)(vh + headBase + (size_t)(kb * 64 + r) * HD + col);
#pragma unroll
            for (int e = 0; e < 8; ++e) Vt[(col + e) * LQ + r] = d[e];
        }
        __syncthreads();

        v8f sc[4];
#pragma unroll
        for (int t = 0; t < 4; ++t)
#pragma unroll
            for (int e = 0; e < 8; ++e) sc[t][e] = 0.0f;
#pragma unroll
        for (int kw = 0; kw < 2; ++kw) {
            v16h a = load_a_frag(&Qs[wv * 16 * LQ + kw * 32], LQ, lane);
#pragma unroll
            for (int t = 0; t < 4; ++t) {
                v16h b = load_b_frag(&Ks[t * 16 * LQ + kw * 32], LQ, lane);
                sc[t] = wmma_f16(a, b, sc[t]);
            }
        }

#pragma unroll
        for (int t = 0; t < 4; ++t) {
#pragma unroll
            for (int j = 0; j < 8; ++j) {
                float s = sc[t][j] * 0.125f;     // 1/sqrt(64)
                int qg = qbase + j + 8 * lh;
                int kg = kb * 64 + t * 16 + ln;
                sc[t][j] = (kg > qg) ? NEG_BIG : s;
            }
        }

#pragma unroll
        for (int j = 0; j < 8; ++j) {
            float mx = fmaxf(fmaxf(sc[0][j], sc[1][j]), fmaxf(sc[2][j], sc[3][j]));
#pragma unroll
            for (int off = 1; off < 16; off <<= 1)
                mx = fmaxf(mx, __shfl_xor(mx, off, 32));
            float mnew  = fmaxf(rM[j], mx);
            float alpha = __expf(rM[j] - mnew);
            rM[j] = mnew;
            float sum = 0.0f;
#pragma unroll
            for (int t = 0; t < 4; ++t) {
                float p = __expf(sc[t][j] - mnew);
                sc[t][j] = p;
                sum += p;
            }
#pragma unroll
            for (int off = 1; off < 16; off <<= 1)
                sum += __shfl_xor(sum, off, 32);
            rL[j] = rL[j] * alpha + sum;
#pragma unroll
            for (int t = 0; t < 4; ++t) o[t][j] *= alpha;
#pragma unroll
            for (int t = 0; t < 4; ++t)
                Pw[(j + 8 * lh) * LQ + t * 16 + ln] = (_Float16)sc[t][j];
        }

#pragma unroll
        for (int kw = 0; kw < 2; ++kw) {
            v16h a = load_a_frag(&Pw[kw * 32], LQ, lane);
#pragma unroll
            for (int t = 0; t < 4; ++t) {
                v16h b = load_b_frag(&Vt[t * 16 * LQ + kw * 32], LQ, lane);
                o[t] = wmma_f16(a, b, o[t]);
            }
        }
    }

    const int b = bz >> 4, h = bz & 15;
#pragma unroll
    for (int t = 0; t < 4; ++t) {
#pragma unroll
        for (int j = 0; j < 8; ++j) {
            int row = j + 8 * lh;
            int sg  = qb * 128 + wv * 16 + row;
            float val = o[t][j] / rL[j];
            ctx[((size_t)b * SEQ + sg) * HID + h * HD + t * 16 + ln] = (_Float16)val;
        }
    }
}

// ---------------- host-side launch ----------------
extern "C" void kernel_launch(void* const* d_in, const int* in_sizes, int n_in,
                              void* d_out, int out_size, void* d_ws, size_t ws_size,
                              hipStream_t stream) {
    (void)in_sizes; (void)n_in; (void)out_size; (void)ws_size;
    const float* x      = (const float*)d_in[0];   // [B,S,HID]
    const float* qkv_w  = (const float*)d_in[1];   // [HID, 3*HID]
    const float* qkv_b  = (const float*)d_in[2];   // [3*HID]
    const float* out_w  = (const float*)d_in[3];   // [HID, HID]
    const float* out_b  = (const float*)d_in[4];   // [HID]
    float* out = (float*)d_out;

    _Float16* qkvwT = (_Float16*)d_ws;                       // [3072][1024]
    _Float16* outwT = qkvwT + (size_t)3 * HID * HID;         // [1024][1024]
    size_t perBuf = (size_t)BATCH * NHEADS * SEQ * HD;       // 8,388,608 halves
    _Float16* qh   = outwT + (size_t)HID * HID;
    _Float16* kh   = qh + perBuf;
    _Float16* vh   = kh + perBuf;
    _Float16* ctxh = vh + perBuf;                            // [8192][1024]

    k_transpose_cvt<<<(3 * HID * HID + 255) / 256, 256, 0, stream>>>(
        qkv_w, qkvwT, HID, 3 * HID);
    k_transpose_cvt<<<(HID * HID + 255) / 256, 256, 0, stream>>>(
        out_w, outwT, HID, HID);

    k_qkv_gemm<<<dim3(3 * HID / 128, MTOT / 128), 256, 0, stream>>>(
        x, qkvwT, qkv_b, qh, kh, vh);

    k_attn<<<dim3(SEQ / 128, BATCH * NHEADS), 256, 0, stream>>>(
        qh, kh, vh, ctxh);

    k_out_gemm<<<dim3(HID / 128, MTOT / 128), 256, 0, stream>>>(
        ctxh, outwT, out_b, out);
}
